// CEBlock_89172110999846
// MI455X (gfx1250) — matrix operand order repaired and checked by reading, not compiled, not run on hardware
//
#include <hip/hip_runtime.h>
#include <hip/hip_bf16.h>

// ---------------------------------------------------------------------------
// CEBlock for MI455X (gfx1250, wave32). All dense math via v_wmma_f32_16x16x32_bf16.
// 64x64 block tile, 4 waves, 32x32 per wave (4 WMMA accumulators), register-
// pipelined global->LDS staging with fully packed bf16 conversion/stores:
//   A: float4 (b128) global loads -> v4bf (b64) LDS stores
//   B: strided b32 global loads   -> v2bf (b32) LDS stores
// ---------------------------------------------------------------------------

typedef __attribute__((ext_vector_type(16))) __bf16 v16bf;
typedef __attribute__((ext_vector_type(8)))  __bf16 v8bf;
typedef __attribute__((ext_vector_type(4)))  __bf16 v4bf;
typedef __attribute__((ext_vector_type(2)))  __bf16 v2bf;
typedef __attribute__((ext_vector_type(8)))  float  v8f;

__device__ __forceinline__ float gelu_exact(float v) {
    return 0.5f * v * (1.0f + erff(v * 0.70710678118654752f));
}

__device__ __forceinline__ v16bf mk16(const __bf16* lo, const __bf16* hi) {
    v8bf l = *(const v8bf*)lo;
    v8bf h = *(const v8bf*)hi;
    v16bf r;
    #pragma unroll
    for (int i = 0; i < 8; ++i) { r[i] = l[i]; r[8 + i] = h[i]; }
    return r;
}

#define BM 64
#define BN 64
#define BKK 32
#define LDP 40   // padded LDS row stride (bf16): 80B -> 16B-aligned rows, conflict-free frag reads

// Generic batched strided GEMM: out = epilogue( A[MxK] @ B[KxN] )
//   A elem:  Ab[m*sAr + k]   (k-stride must be 1 and sAr % 4 == 0; true for all uses here)
//   B elem:  Bb[k*sBr + n*sBc]
//   epilogue: f = acc (+bias[n]); if gelu f=gelu(f); f*=scale; (+res[m*sRr+n]); store
__global__ __launch_bounds__(128)
void gemm_wmma_bf16(const float* __restrict__ A, long long sAr, long long sAbo, long long sAbi,
                    const float* __restrict__ B, long long sBr, long long sBc, long long sBbo, long long sBbi,
                    const float* __restrict__ bias,
                    const float* __restrict__ res, long long sRr,
                    float* __restrict__ out, long long sOr, long long sObo, long long sObi,
                    int M, int N, int K, int binner, float scale, int do_gelu)
{
    __shared__ __bf16 As[BM * LDP];
    __shared__ __bf16 Bs[BN * LDP];

    const int z  = blockIdx.z;
    const int zo = z / binner, zi = z - zo * binner;
    const float* Ab = A + zo * sAbo + zi * sAbi;
    const float* Bb = B + zo * sBbo + zi * sBbi;
    float*       Ob = out + zo * sObo + zi * sObi;

    const int row0 = blockIdx.y * BM;
    const int col0 = blockIdx.x * BN;

    const int tid  = threadIdx.x;
    const int lane = tid & 31;
    const int wave = tid >> 5;
    const int wm   = wave & 1;    // 32-row slice within 64
    const int wn   = wave >> 1;   // 32-col slice within 64
    const int r    = lane & 15;
    const int kg   = lane >> 4;   // 0/1 half-wave

    // ---- hoisted staging addresses (all muls outside the K loop) ----
    // A: thread owns k-quad kq of rows m0a, m0a+16, m0a+32, m0a+48 (float4 loads)
    const int m0a = tid >> 3;           // 0..15
    const int kq  = (tid & 7) * 4;      // k quad
    // B: thread owns column nB, 16 consecutive k at kh..kh+15
    const int kh  = (tid >> 6) * 16;    // 0 or 16
    const int nB  = tid & 63;

    const float* aP = Ab + (long long)(row0 + m0a) * sAr + kq;  // += BKK per step
    const long long aStep = 16 * sAr;
    const float* bP0 = Bb + (long long)kh * sBr + (long long)(col0 + nB) * sBc; // += BKK*sBr
    const long long bStep = 2 * sBr;
    const long long bAdv  = (long long)BKK * sBr;

    __bf16* asW = &As[m0a * LDP + kq];     // += 16*LDP per e (b64 quad stores)
    __bf16* bsW = &Bs[nB * LDP + kh];      // += 2 per e (b32 pair stores)

    v8f acc00 = {}, acc01 = {}, acc10 = {}, acc11 = {};

    float4 ra[4];
    float  rb[16];
    // prologue: fetch first tile into registers
    {
        const float* p = aP;
        #pragma unroll
        for (int e = 0; e < 4; ++e) { ra[e] = *(const float4*)p; p += aStep; }
        const float* q0 = bP0;
        const float* q1 = bP0 + sBr;
        #pragma unroll
        for (int e = 0; e < 8; ++e) { rb[2*e] = *q0; rb[2*e+1] = *q1; q0 += bStep; q1 += bStep; }
        aP += BKK; bP0 += bAdv;
    }

    for (int k0 = 0; k0 < K; k0 += BKK) {
        __syncthreads();   // previous iteration's LDS reads complete
        // ---- convert+store staged registers to LDS (packed) ----
        #pragma unroll
        for (int e = 0; e < 4; ++e) {
            v4bf t;
            t[0] = (__bf16)ra[e].x; t[1] = (__bf16)ra[e].y;
            t[2] = (__bf16)ra[e].z; t[3] = (__bf16)ra[e].w;
            *(v4bf*)(asW + e * 16 * LDP) = t;
        }
        #pragma unroll
        for (int e = 0; e < 8; ++e) {
            v2bf t; t[0] = (__bf16)rb[2*e]; t[1] = (__bf16)rb[2*e+1];
            *(v2bf*)(bsW + 2 * e) = t;
        }
        __syncthreads();

        // ---- issue next tile's global loads (overlap with WMMA) ----
        if (k0 + BKK < K) {
            const float* p = aP;
            #pragma unroll
            for (int e = 0; e < 4; ++e) { ra[e] = *(const float4*)p; p += aStep; }
            const float* q0 = bP0;
            const float* q1 = bP0 + sBr;
            #pragma unroll
            for (int e = 0; e < 8; ++e) { rb[2*e] = *q0; rb[2*e+1] = *q1; q0 += bStep; q1 += bStep; }
            aP += BKK; bP0 += bAdv;
            if (k0 + 2 * BKK < K) __builtin_prefetch(aP, 0, 1);   // global_prefetch_b8
        }

        // ---- fragments: 2 A (rows) x 2 B (cols), ISA wave32 layouts ----
        const __bf16* ap0 = &As[(wm * 32 + r) * LDP];
        const __bf16* ap1 = &As[(wm * 32 + 16 + r) * LDP];
        v16bf a0 = mk16(ap0 + kg * 8, ap0 + 16 + kg * 8);
        v16bf a1 = mk16(ap1 + kg * 8, ap1 + 16 + kg * 8);
        const __bf16* bp0 = &Bs[(wn * 32 + r) * LDP];
        const __bf16* bp1 = &Bs[(wn * 32 + 16 + r) * LDP];
        v16bf b0 = mk16(bp0 + kg * 8, bp0 + 16 + kg * 8);
        v16bf b1 = mk16(bp1 + kg * 8, bp1 + 16 + kg * 8);

        acc00 = __builtin_amdgcn_wmma_f32_16x16x32_bf16(false, a0, false, b0, (short)0, acc00, false, false);
        acc01 = __builtin_amdgcn_wmma_f32_16x16x32_bf16(false, a0, false, b1, (short)0, acc01, false, false);
        acc10 = __builtin_amdgcn_wmma_f32_16x16x32_bf16(false, a1, false, b0, (short)0, acc10, false, false);
        acc11 = __builtin_amdgcn_wmma_f32_16x16x32_bf16(false, a1, false, b1, (short)0, acc11, false, false);
    }

    // ---- epilogue: C layout lane=(N), vgpr v -> M = v + 8*(lane>>4) ----
    const int cn = lane & 15;
    const int mh = lane >> 4;
    auto emit = [&](const v8f& acc, int mbase, int nbase) {
        int gn = nbase + cn;
        float bv = bias ? bias[gn] : 0.f;
        #pragma unroll
        for (int v = 0; v < 8; ++v) {
            int m = mbase + mh * 8 + v;
            float f = acc[v] + bv;
            if (do_gelu) f = gelu_exact(f);
            f *= scale;
            if (res) f += res[(long long)m * sRr + gn];
            Ob[(long long)m * sOr + gn] = f;
        }
    };
    emit(acc00, row0 + wm * 32,      col0 + wn * 32);
    emit(acc01, row0 + wm * 32,      col0 + wn * 32 + 16);
    emit(acc10, row0 + wm * 32 + 16, col0 + wn * 32);
    emit(acc11, row0 + wm * 32 + 16, col0 + wn * 32 + 16);
}

// ---- LayerNorm over C=768: one wave32 per row, 24 elems/lane ----
__global__ __launch_bounds__(256)
void ln768(const float* __restrict__ x, const float* __restrict__ g, const float* __restrict__ b,
           float* __restrict__ out, int rows)
{
    const int lane = threadIdx.x & 31;
    const int row  = blockIdx.x * 8 + (threadIdx.x >> 5);
    if (row >= rows) return;
    const float* xr = x + (long long)row * 768;
    float v[24];
    float s = 0.f;
    #pragma unroll
    for (int i = 0; i < 24; ++i) { v[i] = xr[lane + i * 32]; s += v[i]; }
    #pragma unroll
    for (int o = 16; o > 0; o >>= 1) s += __shfl_xor(s, o, 32);
    float mu = s * (1.f / 768.f);
    float q = 0.f;
    #pragma unroll
    for (int i = 0; i < 24; ++i) { float d = v[i] - mu; q += d * d; }
    #pragma unroll
    for (int o = 16; o > 0; o >>= 1) q += __shfl_xor(q, o, 32);
    float inv = rsqrtf(q * (1.f / 768.f) + 1e-5f);
    float* orow = out + (long long)row * 768;
    #pragma unroll
    for (int i = 0; i < 24; ++i) {
        int c = lane + i * 32;
        orow[c] = (v[i] - mu) * inv * g[c] + b[c];
    }
}

// ---- softmax over rows of width 320, in place: one wave32 per row ----
__global__ __launch_bounds__(256)
void softmax320(float* __restrict__ a, int rows)
{
    const int lane = threadIdx.x & 31;
    const int row  = blockIdx.x * 8 + (threadIdx.x >> 5);
    if (row >= rows) return;
    float* p = a + (long long)row * 320;
    float v[10];
    float mx = -3.4e38f;
    #pragma unroll
    for (int i = 0; i < 10; ++i) { v[i] = p[lane + 32 * i]; mx = fmaxf(mx, v[i]); }
    #pragma unroll
    for (int o = 16; o > 0; o >>= 1) mx = fmaxf(mx, __shfl_xor(mx, o, 32));
    float s = 0.f;
    #pragma unroll
    for (int i = 0; i < 10; ++i) { v[i] = __expf(v[i] - mx); s += v[i]; }
    #pragma unroll
    for (int o = 16; o > 0; o >>= 1) s += __shfl_xor(s, o, 32);
    float inv = 1.f / s;
    #pragma unroll
    for (int i = 0; i < 10; ++i) p[lane + 32 * i] = v[i] * inv;
}

// ---- token importance: mean over 12 heads x first 64 queries of attn[...,64+j] ----
__global__ void attn_token_scores(const float* __restrict__ attn, float* __restrict__ sc)
{
    int idx = blockIdx.x * blockDim.x + threadIdx.x;
    if (idx >= 32 * 256) return;
    int b = idx >> 8, j = idx & 255;
    float s = 0.f;
    for (int h = 0; h < 12; ++h) {
        const float* p = attn + ((long long)(b * 12 + h) * 320) * 320 + 64 + j;
        #pragma unroll 8
        for (int i = 0; i < 64; ++i) s += p[(long long)i * 320];
    }
    sc[idx] = s * (1.f / 768.f);
}

// ---- per-batch stable descending bitonic sort of 256 scores; emit top-k ----
__global__ __launch_bounds__(256)
void topk_sort(const float* __restrict__ sc, const long long* __restrict__ gis,
               int* __restrict__ topk, float* __restrict__ keep_out, float* __restrict__ rem_out)
{
    __shared__ float ks[256];
    __shared__ int   ki[256];
    int b = blockIdx.x, t = threadIdx.x;
    ks[t] = sc[b * 256 + t];
    ki[t] = t;
    __syncthreads();
    for (int k = 2; k <= 256; k <<= 1)
        for (int j = k >> 1; j > 0; j >>= 1) {
            int ixj = t ^ j;
            if (ixj > t) {
                bool up = ((t & k) == 0);
                float sa = ks[t], sb = ks[ixj];
                int ia = ki[t], ib = ki[ixj];
                bool before = (sa > sb) || (sa == sb && ia < ib);  // desc, stable
                if (up != before) { ks[t] = sb; ks[ixj] = sa; ki[t] = ib; ki[ixj] = ia; }
            }
            __syncthreads();
        }
    int idx = ki[t];
    long long gv = gis[b * 256 + idx];
    if (t < 180) { topk[b * 256 + t] = idx; keep_out[b * 180 + t] = (float)gv; }
    else         { rem_out[b * 76 + (t - 180)] = (float)gv; }
}

// ---- build x2[B,244,768]: 64 t-tokens then 180 gathered attentive tokens ----
__global__ void gather_x2(const float* __restrict__ x1, const int* __restrict__ topk,
                          float* __restrict__ x2)
{
    long long idx = (long long)blockIdx.x * blockDim.x + threadIdx.x;
    if (idx >= (long long)32 * 244 * 768) return;
    int c = (int)(idx % 768);
    long long rr = idx / 768;
    int r = (int)(rr % 244);
    int b = (int)(rr / 244);
    int src = (r < 64) ? r : (64 + topk[b * 256 + (r - 64)]);
    x2[idx] = x1[((long long)b * 320 + src) * 768 + c];
}

__global__ void add_inplace(float* __restrict__ dst, const float* __restrict__ src, long long n)
{
    long long i = (long long)blockIdx.x * blockDim.x + threadIdx.x;
    if (i < n) dst[i] += src[i];
}

// ---------------------------------------------------------------------------
extern "C" void kernel_launch(void* const* d_in, const int* in_sizes, int n_in,
                              void* d_out, int out_size, void* d_ws, size_t ws_size,
                              hipStream_t stream) {
    (void)in_sizes; (void)n_in; (void)out_size; (void)ws_size;
    const float*     x      = (const float*)d_in[0];
    const long long* gis    = (const long long*)d_in[1];
    const float* n1g = (const float*)d_in[2],  *n1b = (const float*)d_in[3];
    const float* wqkv = (const float*)d_in[4], *bqkv = (const float*)d_in[5];
    const float* wproj = (const float*)d_in[6], *bproj = (const float*)d_in[7];
    const float* sa1w = (const float*)d_in[8], *sa1b = (const float*)d_in[9];
    const float* sa2w = (const float*)d_in[10], *sa2b = (const float*)d_in[11];
    const float* n2g = (const float*)d_in[12], *n2b = (const float*)d_in[13];
    const float* f1w = (const float*)d_in[14], *f1b = (const float*)d_in[15];
    const float* f2w = (const float*)d_in[16], *f2b = (const float*)d_in[17];
    const float* m1w = (const float*)d_in[18], *m1b = (const float*)d_in[19];
    const float* m2w = (const float*)d_in[20], *m2b = (const float*)d_in[21];

    const int Bn = 32, L = 320, C = 768, H = 12, M1 = Bn * L, M2 = Bn * 244;
    const long long P = (long long)M1 * C;          // 7,864,320
    const long long Q = (long long)M1 * 2304;       // qkv
    const long long A1SZ = (long long)M1 * 192;

    // workspace layout (floats)
    float* W = (float*)d_ws;
    float* w_xn  = W;                         // phase1: xn ; phase2 reuse: x2
    float* w_qkv = W + P;
    float* w_xat = w_qkv + Q;
    float* w_t0  = w_xat + P;
    float* w_a1  = w_t0 + P;
    float* w_x1  = w_a1 + A1SZ;
    float* w_sc  = w_x1 + P;
    int*   w_tk  = (int*)(w_sc + 8192);
    // phase-2 buffers overlay dead phase-1 regions
    float* w_x2  = w_xn;                           // M2*768
    float* w_xn2 = w_qkv;                          // M2*768
    float* w_h1  = w_qkv + (long long)M2 * 768;    // M2*3072
    float* w_m   = w_h1 + (long long)M2 * 3072;    // M2*768
    float* w_a2  = w_m + (long long)M2 * 768;      // M2*192

    // output layout (floats)
    float* o_x    = (float*)d_out;                               // M2*768
    float* o_keep = o_x + (long long)M2 * 768;                   // 32*180
    float* o_rem  = o_keep + 32 * 180;                           // 32*76
    float* o_attn = o_rem + 32 * 76;                             // 32*12*320*320

    const long long ZL = 0;
    dim3 blk(256);
    dim3 gblk(128);

    // 1) LN1
    ln768<<<dim3(M1 / 8), blk, 0, stream>>>(x, n1g, n1b, w_xn, M1);

    // 2) qkv = xn @ wqkv + bqkv          [10240 x 2304 x 768]
    gemm_wmma_bf16<<<dim3(2304 / 64, M1 / 64, 1), gblk, 0, stream>>>(
        w_xn, 768, ZL, ZL,  wqkv, 2304, 1, ZL, ZL,
        bqkv, nullptr, ZL,  w_qkv, 2304, ZL, ZL,
        M1, 2304, 768, 1, 1.0f, 0);

    // 3) scores = q @ k^T * hd^-0.5  per (b,h)  -> attn region (pre-softmax)
    gemm_wmma_bf16<<<dim3(320 / 64, 320 / 64, Bn * H), gblk, 0, stream>>>(
        w_qkv,        2304, (long long)L * 2304, 64,            // A = q
        w_qkv + 768,  1,    2304, (long long)L * 2304, 64,      // B[d][j] = k[j][d]
        nullptr, nullptr, ZL,
        o_attn, 320, (long long)H * L * L, (long long)L * L,
        L, L, 64, H, 0.125f, 0);

    // 4) softmax rows of 320, in place
    softmax320<<<dim3(Bn * H * L / 8), blk, 0, stream>>>(o_attn, Bn * H * L);

    // 5) x_attn_heads = attn @ v  per (b,h) -> [B,L,H,hd] = [10240 x 768]
    gemm_wmma_bf16<<<dim3(64 / 64, 320 / 64, Bn * H), gblk, 0, stream>>>(
        o_attn, 320, (long long)H * L * L, (long long)L * L,
        w_qkv + 1536, 2304, 1, (long long)L * 2304, 64,
        nullptr, nullptr, ZL,
        w_xat, 768, (long long)L * 768, 64,
        L, 64, L, H, 1.0f, 0);

    // 6) t0 = x_attn @ wproj + bproj
    gemm_wmma_bf16<<<dim3(768 / 64, M1 / 64, 1), gblk, 0, stream>>>(
        w_xat, 768, ZL, ZL,  wproj, 768, 1, ZL, ZL,
        bproj, nullptr, ZL,  w_t0, 768, ZL, ZL,
        M1, 768, 768, 1, 1.0f, 0);

    // 7) a1 = gelu(t0 @ sa_fc1 + b)
    gemm_wmma_bf16<<<dim3(192 / 64, M1 / 64, 1), gblk, 0, stream>>>(
        w_t0, 768, ZL, ZL,  sa1w, 192, 1, ZL, ZL,
        sa1b, nullptr, ZL,  w_a1, 192, ZL, ZL,
        M1, 192, 768, 1, 1.0f, 1);

    // 8) t0 += x   (t0 becomes x + x_attn_proj)
    add_inplace<<<dim3((unsigned)((P + 255) / 256)), blk, 0, stream>>>(w_t0, x, P);

    // 9) x1 = (a1 @ sa_fc2 + b) + t0
    gemm_wmma_bf16<<<dim3(768 / 64, M1 / 64, 1), gblk, 0, stream>>>(
        w_a1, 192, ZL, ZL,  sa2w, 768, 1, ZL, ZL,
        sa2b, w_t0, 768,    w_x1, 768, ZL, ZL,
        M1, 768, 192, 1, 1.0f, 0);

    // 10) token scores
    attn_token_scores<<<dim3(32), blk, 0, stream>>>(o_attn, w_sc);

    // 11) top-k sort + index gathers
    topk_sort<<<dim3(Bn), blk, 0, stream>>>(w_sc, gis, w_tk, o_keep, o_rem);

    // 12) x2 = concat(tokens_t, gathered attentive tokens)
    gather_x2<<<dim3((unsigned)(((long long)M2 * 768 + 255) / 256)), blk, 0, stream>>>(
        w_x1, w_tk, w_x2);

    // 13) LN2
    ln768<<<dim3(M2 / 8), blk, 0, stream>>>(w_x2, n2g, n2b, w_xn2, M2);

    // 14) h1 = gelu(xn2 @ fc1 + b)
    gemm_wmma_bf16<<<dim3(3072 / 64, M2 / 64, 1), gblk, 0, stream>>>(
        w_xn2, 768, ZL, ZL,  f1w, 3072, 1, ZL, ZL,
        f1b, nullptr, ZL,    w_h1, 3072, ZL, ZL,
        M2, 3072, 768, 1, 1.0f, 1);

    // 15) m = (h1 @ fc2 + b) + x2
    gemm_wmma_bf16<<<dim3(768 / 64, M2 / 64, 1), gblk, 0, stream>>>(
        w_h1, 3072, ZL, ZL,  f2w, 768, 1, ZL, ZL,
        f2b, w_x2, 768,      w_m, 768, ZL, ZL,
        M2, 768, 3072, 1, 1.0f, 0);

    // 16) a2 = gelu(xn2 @ ma_fc1 + b)
    gemm_wmma_bf16<<<dim3(192 / 64, M2 / 64, 1), gblk, 0, stream>>>(
        w_xn2, 768, ZL, ZL,  m1w, 192, 1, ZL, ZL,
        m1b, nullptr, ZL,    w_a2, 192, ZL, ZL,
        M2, 192, 768, 1, 1.0f, 1);

    // 17) out_x = 0.5*(a2 @ ma_fc2 + b) + m
    gemm_wmma_bf16<<<dim3(768 / 64, M2 / 64, 1), gblk, 0, stream>>>(
        w_a2, 192, ZL, ZL,  m2w, 768, 1, ZL, ZL,
        m2b, w_m, 768,      o_x, 768, ZL, ZL,
        M2, 768, 192, 1, 0.5f, 0);
}
